// MultiInputLSTMCell_23965917512359
// MI455X (gfx1250) — matrix operand order repaired and checked by reading, not compiled, not run on hardware
//
#include <hip/hip_runtime.h>
#include <hip/hip_bf16.h>

#define H 512
#define KROWS 32768
#define MT 128           // c_input rows per workgroup
#define BLOCK 512        // 16 waves of 32

typedef __attribute__((ext_vector_type(16))) __bf16 bf16x16;
typedef __attribute__((ext_vector_type(8)))  float  f32x8;

union Frag { uint4 q[2]; bf16x16 v; };

// Pack two fp32 -> two bf16 (truncating) in ONE v_perm_b32:
// out[15:0] = a[31:16], out[31:16] = b[31:16]
__device__ __forceinline__ unsigned int pk_bf16(float a, float b) {
  return __builtin_amdgcn_perm(__float_as_uint(b), __float_as_uint(a),
                               0x07060302u);
}

__device__ __forceinline__ unsigned short bf16_1(float a) {
  return (unsigned short)(__float_as_uint(a) >> 16);
}

#if __has_builtin(__builtin_amdgcn_s_wait_asynccnt)
#define WAIT_ASYNC0() __builtin_amdgcn_s_wait_asynccnt(0)
#else
#define WAIT_ASYNC0() asm volatile("s_wait_asynccnt 0" ::: "memory")
#endif

// ---------------------------------------------------------------------------
// Setup: gate pre-activations (i,o,g) + alpha_wi GEMVs.  j in [0,3H): gates,
// j in [3H,4H): alpha_wi column.
// ---------------------------------------------------------------------------
__global__ __launch_bounds__(256) void k_gates(
    const float* __restrict__ in_, const float* __restrict__ h0,
    const float* __restrict__ Wih, const float* __restrict__ Whh,
    const float* __restrict__ bias, const float* __restrict__ aWih,
    float* __restrict__ wsD, float* __restrict__ wsO,
    float* __restrict__ wsG, float* __restrict__ wsAwi) {
  int j = blockIdx.x * 256 + threadIdx.x;
  if (j < 3 * H) {
    float acc = bias[j];
    for (int d = 0; d < H; ++d)
      acc += h0[d] * Whh[d * 3 * H + j] + in_[d] * Wih[d * 3 * H + j];
    if (j < H)            wsD[j]         = __expf(1.f / (1.f + __expf(-acc))); // exp(sigmoid(i))
    else if (j < 2 * H)   wsO[j - H]     = 1.f / (1.f + __expf(-acc));         // sigmoid(o)
    else                  wsG[j - 2 * H] = tanhf(acc);                          // tanh(g)
  } else if (j < 4 * H) {
    int h = j - 3 * H;
    float acc = 0.f;
    for (int d = 0; d < H; ++d) acc += in_[d] * aWih[d * H + h];
    wsAwi[h] = acc;
  }
}

// numer accumulator seed = tanh(g) * exp(sigmoid(i))   (softmax row 0)
__global__ void k_init(const float* __restrict__ wsD,
                       const float* __restrict__ wsG, float* __restrict__ wsN) {
  int h = blockIdx.x * 256 + threadIdx.x;
  if (h < H) wsN[h] = wsG[h] * wsD[h];
}

// ---------------------------------------------------------------------------
// Convert alpha_weight_hh (fp32, [k][n] row-major) -> bf16 in FRAGMENT ORDER:
// Bt[kc][ntile][lane][e]  (kc = k/32, ntile = n/16, lane = WMMA lane, e = 0..15)
//   n = ntile*16 + (lane&15),  k = kc*32 + (e>>3)*16 + (lane>>4)*8 + (e&7)
// A 32KB chunk is then a byte-for-byte async copy, and each lane's B fragment
// is 32 contiguous bytes in LDS (conflict-free, zero padding).
// ---------------------------------------------------------------------------
__global__ __launch_bounds__(256) void k_convB(const float* __restrict__ W,
                                               unsigned short* __restrict__ Bt) {
  int o = blockIdx.x * 256 + threadIdx.x;   // 0 .. 262143
  int kc    = o >> 14;                       // chunk (k/32)
  int f     = o & 16383;                     // offset within chunk (u16 units)
  int ntile = f >> 9;                        // 0..31
  int lane  = (f >> 4) & 31;
  int e     = f & 15;
  int n     = (ntile << 4) + (lane & 15);
  int k     = (kc << 5) + ((e >> 3) << 4) + ((lane >> 4) << 3) + (e & 7);
  Bt[o] = bf16_1(W[k * H + n]);
}

// ---------------------------------------------------------------------------
// Main fused kernel: double-buffered GLOBAL_LOAD_ASYNC_TO_LDS_B128 staging of
// fragment-ordered B chunks, WMMA bf16 main loop, fused
// sigmoid->exp->softmax-weighted reduce of c_input down the row axis.
// ---------------------------------------------------------------------------
__global__ __launch_bounds__(BLOCK) void k_main(
    const float* __restrict__ c_input, const unsigned short* __restrict__ Bt,
    const float* __restrict__ wsAwi, float* __restrict__ accD,
    float* __restrict__ accN) {
  __shared__ __align__(16) char pool[65536];   // 2 x 32KB B-chunk buffers

  const int tid  = threadIdx.x;
  const int wave = tid >> 5;
  const int lane = tid & 31;
  const int l15  = lane & 15;
  const int hl   = lane >> 4;
  const int wm   = wave >> 1;            // 0..7  (row group of 16)
  const int wn   = wave & 1;             // 0..1  (column half of 256)
  const int m0   = blockIdx.x * MT;
  const int rowA = m0 + (wm << 4) + l15; // this lane's A-matrix row
  const int khA  = hl << 3;              // A half-select: K +0 / +8

  // Accumulators seeded with alpha_wi[n] (broadcast over M via C layout)
  f32x8 acc[16];
  #pragma unroll
  for (int nt = 0; nt < 16; ++nt) {
    float aw = wsAwi[(wn << 8) + (nt << 4) + l15];
    #pragma unroll
    for (int v = 0; v < 8; ++v) acc[nt][v] = aw;
  }

  const float4* Arow = (const float4*)(c_input + (size_t)rowA * H);
  // Low 32 bits of a flat pointer into LDS are the LDS byte address.
  const unsigned poolBase = (unsigned)(size_t)(&pool[0]);

  // Issue one 32KB contiguous fragment-ordered chunk -> LDS buffer (async).
  auto issue_chunk = [&](int kc, unsigned dstBase) {
    const unsigned long long srcBase =
        (unsigned long long)(const char*)Bt + (unsigned)(kc << 15);
    #pragma unroll
    for (int i = 0; i < 4; ++i) {
      int idx = tid + BLOCK * i;              // 0..2047 (16B units)
      unsigned lds = dstBase + (unsigned)(idx << 4);
      unsigned long long ga = srcBase + (unsigned)(idx << 4);
      asm volatile("global_load_async_to_lds_b128 %0, %1, off"
                   :: "v"(lds), "v"(ga) : "memory");
    }
  };

  issue_chunk(0, poolBase);                   // prologue: chunk 0 -> buf 0

  for (int kc = 0; kc < 16; ++kc) {
    const int kk = kc << 5;
    WAIT_ASYNC0();        // this wave's copies for chunk kc have landed
    __syncthreads();      // all waves' copies visible; prev buffer fully read

    if (kc + 1 < 16)      // overlap next chunk's copy with this chunk's math
      issue_chunk(kc + 1, poolBase + (unsigned)(((kc + 1) & 1) << 15));

    // A fragment: fp32 from global, packed to bf16 in registers.
    // ISA layout: V0-3 K = khA..khA+7, V4-7 K = 16+khA..16+khA+7
    Frag a;
    {
      float4 f0 = Arow[(kk + khA) >> 2];
      float4 f1 = Arow[((kk + khA) >> 2) + 1];
      float4 f2 = Arow[(kk + 16 + khA) >> 2];
      float4 f3 = Arow[((kk + 16 + khA) >> 2) + 1];
      a.q[0] = make_uint4(pk_bf16(f0.x, f0.y), pk_bf16(f0.z, f0.w),
                          pk_bf16(f1.x, f1.y), pk_bf16(f1.z, f1.w));
      a.q[1] = make_uint4(pk_bf16(f2.x, f2.y), pk_bf16(f2.z, f2.w),
                          pk_bf16(f3.x, f3.y), pk_bf16(f3.z, f3.w));
    }

    // Fragment-ordered buffer: lane's B frag = 32 contiguous bytes.
    const char* cur = pool + ((kc & 1) << 15);
    const char* bbase = cur + (size_t)(((wn << 4) << 10) + (lane << 5));
    #pragma unroll
    for (int nt = 0; nt < 16; ++nt) {
      Frag b;
      const uint4* bp = (const uint4*)(bbase + (nt << 10));
      b.q[0] = bp[0];
      b.q[1] = bp[1];
      acc[nt] = __builtin_amdgcn_wmma_f32_16x16x32_bf16(
          false, a.v, false, b.v, (short)0, acc[nt], false, false);
    }
  }

  // Reduction arrays alias the (now idle) staging pool.
  __syncthreads();
  float* redD = (float*)pool;                 // 512 floats
  float* redN = (float*)(pool + 2048);        // 512 floats
  for (int c = tid; c < 512; c += BLOCK) { redD[c] = 0.f; redN[c] = 0.f; }
  __syncthreads();

  // Fused epilogue: e = exp(sigmoid(alpha_pre)); partial softmax sums.
  // C layout: element (M = v + 8*hl, N = l15) per VGPR v.
  #pragma unroll
  for (int nt = 0; nt < 16; ++nt) {
    int n = (wn << 8) + (nt << 4) + l15;
    float d = 0.f, nm = 0.f;
    #pragma unroll
    for (int v = 0; v < 8; ++v) {
      float pre = acc[nt][v];
      float s = 1.f / (1.f + __expf(-pre));
      float e = __expf(s);
      int row = m0 + (wm << 4) + (hl << 3) + v;
      float cv = __builtin_nontemporal_load(&c_input[(size_t)row * H + n]);
      d  += e;
      nm += cv * e;
    }
    d  += __shfl_xor(d, 16, 32);     // combine the two lane-halves (same N)
    nm += __shfl_xor(nm, 16, 32);
    if (hl == 0) {
      atomicAdd(&redD[n], d);        // ds_add_f32
      atomicAdd(&redN[n], nm);
    }
  }
  __syncthreads();
  for (int c = tid; c < 512; c += BLOCK) {
    unsafeAtomicAdd(&accD[c], redD[c]);   // global_atomic_add_f32
    unsafeAtomicAdd(&accN[c], redN[c]);
  }
}

// ---------------------------------------------------------------------------
// Final: c1 = numer/(denom+1e-12); h1 = o * tanh(c1).  out = [h1 | c1].
// ---------------------------------------------------------------------------
__global__ void k_final(const float* __restrict__ accD,
                        const float* __restrict__ accN,
                        const float* __restrict__ wsO, float* __restrict__ out) {
  int h = blockIdx.x * 256 + threadIdx.x;
  if (h < H) {
    float c1 = accN[h] / (accD[h] + 1e-12f);
    float h1 = wsO[h] * tanhf(c1);
    out[h]     = h1;
    out[H + h] = c1;
  }
}

extern "C" void kernel_launch(void* const* d_in, const int* in_sizes, int n_in,
                              void* d_out, int out_size, void* d_ws, size_t ws_size,
                              hipStream_t stream) {
  (void)in_sizes; (void)n_in; (void)out_size; (void)ws_size;
  const float* input_    = (const float*)d_in[0];
  const float* c_input   = (const float*)d_in[1];
  const float* h_0       = (const float*)d_in[2];
  /* d_in[3] = c_0 : unused by the non-empty-children branch */
  const float* weight_ih = (const float*)d_in[4];
  const float* weight_hh = (const float*)d_in[5];
  const float* bias      = (const float*)d_in[6];
  const float* aWih      = (const float*)d_in[7];
  const float* aWhh      = (const float*)d_in[8];

  float* ws    = (float*)d_ws;
  float* wsD   = ws;            // denom accumulator (seeded with exp(sigmoid(i)))
  float* wsN   = ws + 512;      // numer accumulator (seeded with g*exp(sigmoid(i)))
  float* wsO   = ws + 1024;     // sigmoid(o)
  float* wsAwi = ws + 1536;     // input_ @ alpha_weight_ih
  float* wsG   = ws + 2048;     // tanh(g) temp
  unsigned short* Bt = (unsigned short*)(ws + 2560);  // bf16 B, 512KB, 16B-aligned

  k_gates<<<8, 256, 0, stream>>>(input_, h_0, weight_ih, weight_hh, bias, aWih,
                                 wsD, wsO, wsG, wsAwi);
  k_init<<<2, 256, 0, stream>>>(wsD, wsG, wsN);
  k_convB<<<1024, 256, 0, stream>>>(aWhh, Bt);
  k_main<<<KROWS / MT, BLOCK, 0, stream>>>(c_input, Bt, wsAwi, wsD, wsN);
  k_final<<<2, 256, 0, stream>>>(wsD, wsN, wsO, (float*)d_out);
}
